// BarrierNet_31104153158091
// MI455X (gfx1250) — matrix operand order repaired and checked by reading, not compiled, not run on hardware
//
#include <hip/hip_runtime.h>
#include <hip/hip_bf16.h>

typedef __attribute__((ext_vector_type(16))) _Float16 v16h;
typedef __attribute__((ext_vector_type(8)))  float    v8f;
typedef __attribute__((ext_vector_type(2)))  _Float16 h2;

#define NB  524288
#define WPB 4                 // waves per block
#define TPB (WPB * 32)
#define STR 36                // padded LDS row stride (dwords), conflict-free

__device__ __forceinline__ h2 relu2(h2 v) {
  // single v_pk_max_num_f16
  h2 z = {};
  return __builtin_elementwise_max(v, z);
}

__device__ __forceinline__ h2 dup16(float x) {
  _Float16 h = (_Float16)x;
  h2 r; r.x = h; r.y = h; return r;
}

__device__ __forceinline__ void st4(float* p, float a, float b, float c, float d) {
  float4 v; v.x = a; v.y = b; v.z = c; v.w = d;
  *(float4*)p = v;
}

__global__ __launch_bounds__(TPB) void barriernet_kernel(
    const float* __restrict__ x,   const float* __restrict__ mean, const float* __restrict__ stdv_,
    const float* __restrict__ W1,  const float* __restrict__ b1,
    const float* __restrict__ W21, const float* __restrict__ b21,
    const float* __restrict__ W22, const float* __restrict__ b22,
    const float* __restrict__ W31, const float* __restrict__ b31,
    const float* __restrict__ W32, const float* __restrict__ b32,
    float* __restrict__ out)
{
  __shared__ unsigned ldsW1[64 * 8];               // 64 k-pair records * 32B
  __shared__ float    stage[WPB][2][32 * STR];     // per-wave [matrix][n][m] staging

  const int tid  = threadIdx.x;
  const int lane = tid & 31;
  const int wv   = tid >> 5;
  const int nn   = lane & 15;
  const int hi   = lane >> 4;                      // 0: lanes 0-15, 1: lanes 16-31

  // ---- one-time: pack W1 (+b1) as f16 pairs into LDS records ----
  if (tid < 64) {
    int p = tid, k0 = 2 * p;
    unsigned* rec = &ldsW1[p * 8];
    #pragma unroll
    for (int f = 0; f < 5; ++f) {
      h2 w; w.x = (_Float16)W1[k0 * 5 + f]; w.y = (_Float16)W1[(k0 + 1) * 5 + f];
      rec[f] = __builtin_bit_cast(unsigned, w);
    }
    h2 bb; bb.x = (_Float16)b1[k0]; bb.y = (_Float16)b1[k0 + 1];
    rec[5] = __builtin_bit_cast(unsigned, bb);
    rec[6] = 0u; rec[7] = 0u;
  }

  // ---- one-time: loop-invariant B fragments for W21 / W22 (f16, VGPR-resident) ----
  const int kb = hi ? 16 : 0;                      // B layout: lanes 16-31 hold K+16
  v16h B21f[4][2], B22f[4][2];
  #pragma unroll
  for (int c = 0; c < 4; ++c) {
    #pragma unroll
    for (int t = 0; t < 2; ++t) {
      int n = nn + 16 * t;
      const float* w21p = W21 + n * 128 + c * 32 + kb;
      const float* w22p = W22 + n * 128 + c * 32 + kb;
      v16h f21, f22;
      #pragma unroll
      for (int e = 0; e < 16; ++e) {
        f21[e] = (_Float16)w21p[e];
        f22[e] = (_Float16)w22p[e];
      }
      B21f[c][t] = f21; B22f[c][t] = f22;
    }
  }

  // per-lane epilogue constants (n is lane-fixed in the C/D layout)
  const float b21v[2] = { b21[nn], b21[nn + 16] };
  const float b22v[2] = { b22[nn], b22[nn + 16] };
  const float w31v[2] = { W31[nn], W31[nn + 16] };
  const float w32v[2] = { W32[nn], W32[nn + 16] };
  const float b31v = b31[0], b32v = b32[0];
  float mv[5], sv[5];
  #pragma unroll
  for (int f = 0; f < 5; ++f) { mv[f] = mean[f]; sv[f] = stdv_[f]; }

  __syncthreads();                                 // cross-wave: W1 LDS records ready

  float* st21 = stage[wv][0];
  float* st22 = stage[wv][1];

  const int stride = gridDim.x * TPB;
  for (int s = blockIdx.x * TPB + tid; s < NB; s += stride) {
    // this lane's input row (also reused for the de-normalized x0 epilogue)
    float xr[5];
    #pragma unroll
    for (int f = 0; f < 5; ++f) xr[f] = x[s * 5 + f];

    // broadcast packed-f16 features for this wave's two 16-row M-tiles
    h2 xt0[5], xt1[5];
    #pragma unroll
    for (int f = 0; f < 5; ++f) {
      xt0[f] = dup16(__shfl(xr[f], nn, 32));
      xt1[f] = dup16(__shfl(xr[f], nn + 16, 32));
    }

    v8f z = {};
    v8f a21_00 = z, a21_01 = z, a21_10 = z, a21_11 = z;   // [mtile][ntile] for W21 path
    v8f a22_00 = z, a22_01 = z, a22_10 = z, a22_11 = z;   // [mtile][ntile] for W22 path

    #pragma unroll
    for (int c = 0; c < 4; ++c) {
      // build layer-1 outputs directly in WMMA A-fragment layout (K chunk c)
      v16h A0, A1;
      #pragma unroll
      for (int j = 0; j < 8; ++j) {
        // A layout: VGPR j holds K pair k0,k0+1 with k0 = 32c + 16*hi + 2j
        int p = 16 * c + 8 * hi + j;
        const uint4* rp = (const uint4*)&ldsW1[p * 8];
        uint4 q0 = rp[0];
        uint4 q1 = rp[1];
        h2 w0 = __builtin_bit_cast(h2, q0.x);
        h2 w1 = __builtin_bit_cast(h2, q0.y);
        h2 w2 = __builtin_bit_cast(h2, q0.z);
        h2 w3 = __builtin_bit_cast(h2, q0.w);
        h2 w4 = __builtin_bit_cast(h2, q1.x);
        h2 bb = __builtin_bit_cast(h2, q1.y);
        h2 a0 = bb, a1 = bb;
        a0 += xt0[0] * w0; a0 += xt0[1] * w1; a0 += xt0[2] * w2;
        a0 += xt0[3] * w3; a0 += xt0[4] * w4;
        a1 += xt1[0] * w0; a1 += xt1[1] * w1; a1 += xt1[2] * w2;
        a1 += xt1[3] * w3; a1 += xt1[4] * w4;
        a0 = relu2(a0); a1 = relu2(a1);
        A0[2 * j] = a0.x; A0[2 * j + 1] = a0.y;
        A1[2 * j] = a1.x; A1[2 * j + 1] = a1.y;
      }
      a21_00 = __builtin_amdgcn_wmma_f32_16x16x32_f16(false, A0, false, B21f[c][0], (short)0, a21_00, false, false);
      a21_01 = __builtin_amdgcn_wmma_f32_16x16x32_f16(false, A0, false, B21f[c][1], (short)0, a21_01, false, false);
      a22_00 = __builtin_amdgcn_wmma_f32_16x16x32_f16(false, A0, false, B22f[c][0], (short)0, a22_00, false, false);
      a22_01 = __builtin_amdgcn_wmma_f32_16x16x32_f16(false, A0, false, B22f[c][1], (short)0, a22_01, false, false);
      a21_10 = __builtin_amdgcn_wmma_f32_16x16x32_f16(false, A1, false, B21f[c][0], (short)0, a21_10, false, false);
      a21_11 = __builtin_amdgcn_wmma_f32_16x16x32_f16(false, A1, false, B21f[c][1], (short)0, a21_11, false, false);
      a22_10 = __builtin_amdgcn_wmma_f32_16x16x32_f16(false, A1, false, B22f[c][0], (short)0, a22_10, false, false);
      a22_11 = __builtin_amdgcn_wmma_f32_16x16x32_f16(false, A1, false, B22f[c][1], (short)0, a22_11, false, false);
    }

    // bias + relu + fold layer-3 weight, stage columns [n][m] to LDS
    #pragma unroll
    for (int t = 0; t < 2; ++t) {
      int n = nn + 16 * t;
      float* d1 = &st21[n * STR + 8 * hi];   // mtile0 rows: m = 8*hi + r
      float* d2 = &st22[n * STR + 8 * hi];
      v8f a1a = (t == 0) ? a21_00 : a21_01;
      v8f a1b = (t == 0) ? a21_10 : a21_11;  // mtile1 rows: m = 16 + 8*hi + r
      v8f a2a = (t == 0) ? a22_00 : a22_01;
      v8f a2b = (t == 0) ? a22_10 : a22_11;
      float t1[8], t2[8], t3[8], t4[8];
      #pragma unroll
      for (int r = 0; r < 8; ++r) {
        t1[r] = fmaxf(a1a[r] + b21v[t], 0.f) * w31v[t];
        t2[r] = fmaxf(a1b[r] + b21v[t], 0.f) * w31v[t];
        t3[r] = fmaxf(a2a[r] + b22v[t], 0.f) * w32v[t];
        t4[r] = fmaxf(a2b[r] + b22v[t], 0.f) * w32v[t];
      }
      st4(d1 + 0,  t1[0], t1[1], t1[2], t1[3]);
      st4(d1 + 4,  t1[4], t1[5], t1[6], t1[7]);
      st4(d1 + 16, t2[0], t2[1], t2[2], t2[3]);
      st4(d1 + 20, t2[4], t2[5], t2[6], t2[7]);
      st4(d2 + 0,  t3[0], t3[1], t3[2], t3[3]);
      st4(d2 + 4,  t3[4], t3[5], t3[6], t3[7]);
      st4(d2 + 16, t4[0], t4[1], t4[2], t4[3]);
      st4(d2 + 20, t4[4], t4[5], t4[6], t4[7]);
    }

    // staging traffic is intra-wave only (per-wave region, DS ops are in order
    // per wave) -> compiler-only barrier, no hardware sync needed
    __builtin_amdgcn_wave_barrier();

    // each lane owns its own sample (m == lane): sum the 32 pre-scaled columns
    float sum1 = 0.f, sum2 = 0.f;
    #pragma unroll
    for (int n = 0; n < 32; ++n) {
      sum1 += st21[n * STR + lane];
      sum2 += st22[n * STR + lane];
    }
    float x31  = sum1 + b31v;                       // QP linear term p
    float tl   = sum2 + b32v;
    float x32  = 4.f / (1.f + __expf(-tl));         // learned penalty
    float x0[5];
    #pragma unroll
    for (int f = 0; f < 5; ++f) x0[f] = xr[f] * sv[f] + mv[f];
    float hh = (x0[1] - x0[3]) + x32 * (x0[0] - x0[2] - 1.8f * x0[3]);
    float uu = -x31;
    float u  = (1.8f * uu <= hh) ? uu : hh * (1.0f / 1.8f);
    out[s] = u;

    __builtin_amdgcn_wave_barrier();               // don't let next iter's stores move up
  }
}

extern "C" void kernel_launch(void* const* d_in, const int* in_sizes, int n_in,
                              void* d_out, int out_size, void* d_ws, size_t ws_size,
                              hipStream_t stream) {
  (void)in_sizes; (void)n_in; (void)out_size; (void)d_ws; (void)ws_size;
  const float* x    = (const float*)d_in[0];
  const float* mean = (const float*)d_in[1];
  const float* stdv = (const float*)d_in[2];
  const float* W1   = (const float*)d_in[3];
  const float* b1   = (const float*)d_in[4];
  const float* W21  = (const float*)d_in[5];
  const float* b21  = (const float*)d_in[6];
  const float* W22  = (const float*)d_in[7];
  const float* b22  = (const float*)d_in[8];
  const float* W31  = (const float*)d_in[9];
  const float* b31  = (const float*)d_in[10];
  const float* W32  = (const float*)d_in[11];
  const float* b32  = (const float*)d_in[12];
  float* out = (float*)d_out;

  // 1024 blocks * 128 threads = 4096 waves; 32 samples/wave/iter -> 4 grid-stride iters
  dim3 grid(1024), block(TPB);
  hipLaunchKernelGGL(barriernet_kernel, grid, block, 0, stream,
                     x, mean, stdv, W1, b1, W21, b21, W22, b22, W31, b31, W32, b32, out);
}
